// GNNMessagePassing_18416819765909
// MI455X (gfx1250) — matrix operand order repaired and checked by reading, not compiled, not run on hardware
//
#include <hip/hip_runtime.h>
#include <hip/hip_bf16.h>

typedef __attribute__((ext_vector_type(16))) _Float16 v16h;
typedef __attribute__((ext_vector_type(8)))  _Float16 v8h;
typedef __attribute__((ext_vector_type(8)))  float    v8f;

#define IN_DIM 256
#define HID    64
#define HEADS  4

// ---- monotone float <-> uint encoding for atomicMax over signed floats ----
__device__ __forceinline__ unsigned fenc(float f) {
    unsigned u = __float_as_uint(f);
    return (u & 0x80000000u) ? ~u : (u | 0x80000000u);
}
__device__ __forceinline__ float fdec(unsigned u) {
    unsigned v = (u & 0x80000000u) ? (u & 0x7FFFFFFFu) : ~u;
    return __uint_as_float(v);
}

// ---------------- enh = nf + 0.1*query, cast to f16 ----------------
__global__ void prep_enh(const float* __restrict__ nf, const float* __restrict__ qe,
                         _Float16* __restrict__ enh, int total) {
    int idx = blockIdx.x * blockDim.x + threadIdx.x;
    if (idx >= total) return;
    enh[idx] = (_Float16)(nf[idx] + 0.1f * qe[idx & (IN_DIM - 1)]);
}

// ---------------- pack weights into WMMA B-fragment layout ----------------
// wpk[kk][ntile][lane][i] = W[K][N], K = kk*32 + (i>>3)*16 + (lane>>4)*8 + (i&7), N = ntile*16 + (lane&15)
__global__ void pack_w(const float* __restrict__ Wq, const float* __restrict__ Wk,
                       const float* __restrict__ Wv, const float* __restrict__ Wo,
                       _Float16* __restrict__ wpk) {
    int t = blockIdx.x * blockDim.x + threadIdx.x;       // 0..65535
    const float* src;
    switch (blockIdx.y) {
        case 0: src = Wq; break;
        case 1: src = Wk; break;
        case 2: src = Wv; break;
        default: src = Wo; break;
    }
    int i    = t & 15;
    int lane = (t >> 4) & 31;
    int nt   = (t >> 9) & 15;
    int kk   = t >> 13;
    int K    = kk * 32 + (i >> 3) * 16 + (lane >> 4) * 8 + (i & 7);
    int Ncol = nt * 16 + (lane & 15);
    wpk[(size_t)blockIdx.y * 65536 + t] = (_Float16)src[K * 256 + Ncol];
}

// ---------------- QKV GEMM: [N,256] x [256,256] via v_wmma_f32_16x16x32_f16 --------
__global__ void gemm_qkv(const _Float16* __restrict__ enh,
                         const _Float16* __restrict__ wpk_base,
                         const float* __restrict__ bq, const float* __restrict__ bk,
                         const float* __restrict__ bv,
                         _Float16* __restrict__ qkv, int N) {
    __shared__ __align__(16) _Float16 Atile[16 * 256];
    const int z = blockIdx.z;
    const _Float16* wpk  = wpk_base + (size_t)z * 65536;
    const float*    bias = (z == 0) ? bq : ((z == 1) ? bk : bv);
    _Float16*       outp = qkv + (size_t)z * (size_t)N * IN_DIM;
    const int m0 = blockIdx.x * 16;

    {   // stage A tile (16 rows x 256 halves = 8 KB) into LDS
        const uint4* src = (const uint4*)(enh + (size_t)m0 * IN_DIM);
        uint4* dst = (uint4*)Atile;
        for (int i = threadIdx.x; i < 512; i += blockDim.x) dst[i] = src[i];
    }
    __syncthreads();

    const int lane = threadIdx.x & 31;
    const int wave = threadIdx.x >> 5;
    const int row  = lane & 15;
    const int g    = lane >> 4;
    const int col  = lane & 15;

    v8f acc[4];
#pragma unroll
    for (int j = 0; j < 4; ++j) {
        float b = bias[(wave * 4 + j) * 16 + col];
#pragma unroll
        for (int r = 0; r < 8; ++r) acc[j][r] = b;
    }

#pragma unroll
    for (int kk = 0; kk < 8; ++kk) {
        v16h a;
        const int K0 = kk * 32;
#pragma unroll
        for (int i = 0; i < 8; ++i) a[i] = Atile[row * 256 + K0 + g * 8 + i];
#pragma unroll
        for (int i = 0; i < 8; ++i) a[8 + i] = Atile[row * 256 + K0 + 16 + g * 8 + i];
#pragma unroll
        for (int j = 0; j < 4; ++j) {
            const int ntile = wave * 4 + j;
            v16h b = *((const v16h*)wpk + ((kk * 16 + ntile) * 32 + lane));
            acc[j] = __builtin_amdgcn_wmma_f32_16x16x32_f16(
                false, a, false, b, (short)0, acc[j], false, false);
        }
    }

#pragma unroll
    for (int j = 0; j < 4; ++j) {
        const int c = (wave * 4 + j) * 16 + col;
#pragma unroll
        for (int r = 0; r < 8; ++r) {
            const int m = g * 8 + r;
            outp[(size_t)(m0 + m) * IN_DIM + c] = (_Float16)acc[j][r];
        }
    }
}

// ---------------- edge pass A: scores + segment max (b128 gathers) ----------------
__global__ void edge_scores(const int* __restrict__ ei, const int* __restrict__ erel,
                            const _Float16* __restrict__ Qh, const _Float16* __restrict__ Kh,
                            const float* __restrict__ relw,
                            float* __restrict__ scores, unsigned* __restrict__ smax, int E) {
    int t = blockIdx.x * blockDim.x + threadIdx.x;
    if (t >= E * HEADS) return;
    int e = t >> 2, h = t & 3;
    int s = ei[e], d = ei[E + e], r = erel[e];
    const v8h*    qv = (const v8h*)(Qh + (size_t)d * IN_DIM + h * HID);
    const v8h*    kv = (const v8h*)(Kh + (size_t)s * IN_DIM + h * HID);
    const float4* rv = (const float4*)(relw + r * HID);
    float acc = 0.f;
#pragma unroll
    for (int j = 0; j < 8; ++j) {
        v8h q8 = qv[j];
        v8h k8 = kv[j];
        float4 r0 = rv[2 * j];
        float4 r1 = rv[2 * j + 1];
        float rr[8] = {r0.x, r0.y, r0.z, r0.w, r1.x, r1.y, r1.z, r1.w};
#pragma unroll
        for (int i = 0; i < 8; ++i)
            acc += (float)q8[i] * ((float)k8[i] + 0.2f * rr[i]);
    }
    float sc = acc * 0.125f;                  // 1/sqrt(64)
    scores[t] = sc;
    atomicMax(&smax[d * HEADS + h], fenc(sc));
}

// ---------------- edge pass B: exp, denom, degree ----------------
__global__ void edge_exp(const int* __restrict__ ei, float* __restrict__ scores,
                         const unsigned* __restrict__ smax, float* __restrict__ denom,
                         float* __restrict__ deg, int E) {
    int t = blockIdx.x * blockDim.x + threadIdx.x;
    if (t >= E * HEADS) return;
    int e = t >> 2, h = t & 3;
    int d = ei[E + e];
    float ex = __expf(scores[t] - fdec(smax[d * HEADS + h]));
    scores[t] = ex;
    atomicAdd(&denom[d * HEADS + h], ex);
    if (h == 0) atomicAdd(&deg[d], 1.0f);
}

// ---------------- edge pass C: weighted V scatter-add (8 threads / (e,h)) ----------
__global__ void edge_agg(const int* __restrict__ ei, const float* __restrict__ scores,
                         const float* __restrict__ denom, const _Float16* __restrict__ Vh,
                         float* __restrict__ agg, int E) {
    int t = blockIdx.x * blockDim.x + threadIdx.x;
    if (t >= E * HEADS * 8) return;
    int c = t & 7;              // 8-dim chunk
    int h = (t >> 3) & 3;
    int e = t >> 5;
    int s = ei[e], d = ei[E + e];
    float w = scores[(e << 2) | h] / denom[d * HEADS + h];
    v8h v8 = *((const v8h*)(Vh + (size_t)s * IN_DIM + h * HID) + c);
    float* ag = agg + (size_t)d * IN_DIM + h * HID + c * 8;
#pragma unroll
    for (int i = 0; i < 8; ++i)
        atomicAdd(&ag[i], w * (float)v8[i]);
}

// ---------------- msgs = deg>0 ? agg : V, cast to f16 ----------------
__global__ void finalize_msgs(const float* __restrict__ deg, const float* __restrict__ agg,
                              const _Float16* __restrict__ Vh, _Float16* __restrict__ msgsh,
                              int total) {
    int idx = blockIdx.x * blockDim.x + threadIdx.x;
    if (idx >= total) return;
    int n = idx >> 8;
    float m = (deg[n] > 0.f) ? agg[idx] : (float)Vh[idx];
    msgsh[idx] = (_Float16)m;
}

// ---------------- output GEMM fused with residual + LayerNorm ----------------
__global__ void gemm_out_ln(const _Float16* __restrict__ msgsh,
                            const _Float16* __restrict__ wpk,
                            const float* __restrict__ bo, const float* __restrict__ nf,
                            const float* __restrict__ gamma, const float* __restrict__ beta,
                            float* __restrict__ out, int N) {
    __shared__ __align__(16) _Float16 Atile[16 * 256];
    __shared__ float Ctile[16 * 256];
    __shared__ float red[16][8], red2[16][8];
    __shared__ float mu_s[16], rstd_s[16];
    const int m0 = blockIdx.x * 16;

    {
        const uint4* src = (const uint4*)(msgsh + (size_t)m0 * IN_DIM);
        uint4* dst = (uint4*)Atile;
        for (int i = threadIdx.x; i < 512; i += blockDim.x) dst[i] = src[i];
    }
    __syncthreads();

    const int lane = threadIdx.x & 31;
    const int wave = threadIdx.x >> 5;
    const int row  = lane & 15;
    const int g    = lane >> 4;
    const int col  = lane & 15;

    v8f acc[4];
#pragma unroll
    for (int j = 0; j < 4; ++j) {
        float b = bo[(wave * 4 + j) * 16 + col];
#pragma unroll
        for (int r = 0; r < 8; ++r) acc[j][r] = b;
    }

#pragma unroll
    for (int kk = 0; kk < 8; ++kk) {
        v16h a;
        const int K0 = kk * 32;
#pragma unroll
        for (int i = 0; i < 8; ++i) a[i] = Atile[row * 256 + K0 + g * 8 + i];
#pragma unroll
        for (int i = 0; i < 8; ++i) a[8 + i] = Atile[row * 256 + K0 + 16 + g * 8 + i];
#pragma unroll
        for (int j = 0; j < 4; ++j) {
            const int ntile = wave * 4 + j;
            v16h b = *((const v16h*)wpk + ((kk * 16 + ntile) * 32 + lane));
            acc[j] = __builtin_amdgcn_wmma_f32_16x16x32_f16(
                false, a, false, b, (short)0, acc[j], false, false);
        }
    }

#pragma unroll
    for (int j = 0; j < 4; ++j) {
        const int c = (wave * 4 + j) * 16 + col;
#pragma unroll
        for (int r = 0; r < 8; ++r)
            Ctile[(g * 8 + r) * 256 + c] = acc[j][r];
    }
    __syncthreads();

    // residual + LayerNorm over each of the 16 rows (8 threads per row)
    const int tid = threadIdx.x;
    const int rr  = tid >> 3;
    const int sub = tid & 7;
    float s = 0.f, ss = 0.f;
    for (int i = sub; i < 256; i += 8) {
        float x = Ctile[rr * 256 + i] + nf[(size_t)(m0 + rr) * IN_DIM + i];
        Ctile[rr * 256 + i] = x;
        s += x; ss += x * x;
    }
    red[rr][sub] = s; red2[rr][sub] = ss;
    __syncthreads();
    if (sub == 0) {
        float S = 0.f, SS = 0.f;
#pragma unroll
        for (int j = 0; j < 8; ++j) { S += red[rr][j]; SS += red2[rr][j]; }
        float mu  = S * (1.0f / 256.0f);
        float var = SS * (1.0f / 256.0f) - mu * mu;
        mu_s[rr]   = mu;
        rstd_s[rr] = rsqrtf(var + 1e-5f);
    }
    __syncthreads();
    for (int i = sub; i < 256; i += 8) {
        float x = Ctile[rr * 256 + i];
        out[(size_t)(m0 + rr) * IN_DIM + i] = (x - mu_s[rr]) * rstd_s[rr] * gamma[i] + beta[i];
    }
}

extern "C" void kernel_launch(void* const* d_in, const int* in_sizes, int n_in,
                              void* d_out, int out_size, void* d_ws, size_t ws_size,
                              hipStream_t stream) {
    const float* nf    = (const float*)d_in[0];
    const int*   ei    = (const int*)d_in[1];
    const int*   erel  = (const int*)d_in[2];
    const float* qemb  = (const float*)d_in[3];
    const float* Wq    = (const float*)d_in[4];
    const float* bq    = (const float*)d_in[5];
    const float* Wk    = (const float*)d_in[6];
    const float* bk    = (const float*)d_in[7];
    const float* Wv    = (const float*)d_in[8];
    const float* bv    = (const float*)d_in[9];
    const float* relw  = (const float*)d_in[10];
    const float* Wo    = (const float*)d_in[11];
    const float* bo    = (const float*)d_in[12];
    const float* gamma = (const float*)d_in[13];
    const float* beta  = (const float*)d_in[14];
    float* out = (float*)d_out;

    const int N = in_sizes[0] / IN_DIM;
    const int E = in_sizes[2];

    char* p = (char*)d_ws;
    auto carve = [&](size_t bytes) -> char* {
        char* r = p;
        p += (bytes + 255) & ~(size_t)255;
        return r;
    };
    _Float16* enh   = (_Float16*)carve((size_t)N * IN_DIM * 2);
    _Float16* wpk   = (_Float16*)carve((size_t)4 * 65536 * 2);   // Wq,Wk,Wv,Wo fragments
    _Float16* qkv   = (_Float16*)carve((size_t)3 * N * IN_DIM * 2);
    float*    scr   = (float*)   carve((size_t)E * HEADS * 4);
    unsigned* smax  = (unsigned*)carve((size_t)N * HEADS * 4);
    float*    denom = (float*)   carve((size_t)N * HEADS * 4);
    float*    deg   = (float*)   carve((size_t)N * 4);
    float*    agg   = (float*)   carve((size_t)N * IN_DIM * 4);
    _Float16* msgsh = (_Float16*)carve((size_t)N * IN_DIM * 2);

    _Float16* Qh = qkv;
    _Float16* Kh = qkv + (size_t)N * IN_DIM;
    _Float16* Vh = qkv + (size_t)2 * N * IN_DIM;

    hipMemsetAsync(smax,  0, (size_t)N * HEADS * 4, stream);   // 0u == min of fenc-order
    hipMemsetAsync(denom, 0, (size_t)N * HEADS * 4, stream);
    hipMemsetAsync(deg,   0, (size_t)N * 4, stream);
    hipMemsetAsync(agg,   0, (size_t)N * IN_DIM * 4, stream);

    const int totNF = N * IN_DIM;
    prep_enh<<<(totNF + 255) / 256, 256, 0, stream>>>(nf, qemb, enh, totNF);
    pack_w<<<dim3(65536 / 256, 4), 256, 0, stream>>>(Wq, Wk, Wv, Wo, wpk);
    gemm_qkv<<<dim3(N / 16, 1, 3), 128, 0, stream>>>(enh, wpk, bq, bk, bv, qkv, N);

    const int totEH = E * HEADS;
    edge_scores<<<(totEH + 255) / 256, 256, 0, stream>>>(ei, erel, Qh, Kh, relw, scr, smax, E);
    edge_exp<<<(totEH + 255) / 256, 256, 0, stream>>>(ei, scr, smax, denom, deg, E);
    const int totEC = E * HEADS * 8;
    edge_agg<<<(totEC + 255) / 256, 256, 0, stream>>>(ei, scr, denom, Vh, agg, E);

    finalize_msgs<<<(totNF + 255) / 256, 256, 0, stream>>>(deg, agg, Vh, msgsh, totNF);
    gemm_out_ln<<<N / 16, 128, 0, stream>>>(msgsh, wpk + (size_t)3 * 65536, bo, nf,
                                            gamma, beta, out, N);
}
